// QResBlockWithBN_32736240730333
// MI455X (gfx1250) — 8-bit matrix operand order repaired and checked by reading, compiled, not run on hardware
//
#include <hip/hip_runtime.h>
#include <hip/hip_bf16.h>
#include <stdint.h>

typedef __attribute__((ext_vector_type(8))) int v8i;

#define QMAXF    127.0f
#define IMG_W    192
#define IMG_HW   36864L
#define NPIX     294912L
#define NELEM    18874368L
#define SEGPIX   64
#define NSEG     4
#define NCBLK    1152
#define TCOLS    66
#define TCHUNK   792
#define BN_EPS   1e-5f

#define OFF_QW1  0UL
#define OFF_QW2  36864UL
#define OFF_SW   73728UL
#define OFF_ZP   73792UL
#define OFF_BNP1 73856UL
#define OFF_BNP2 74368UL
#define OFF_PSUM 74880UL
#define OFF_PSQ  (OFF_PSUM + 294912UL)
#define OFF_X1   665600UL
#define OFF_Y1   (OFF_X1 + 18874368UL)
#define OFF_X2   (OFF_Y1 + 75497472UL)
#define OFF_Y2   (OFF_X2 + 18874368UL)

__device__ __forceinline__ int swz(int c) {
  return ((c >> 3) & 1) * 32 + (c >> 4) * 8 + (c & 7);
}

__global__ void k_prep(const float* __restrict__ w1, const float* __restrict__ w2,
                       const float* bits, const float* f, const float* bits_out,
                       char* __restrict__ ws, float* __restrict__ out_tail) {
  __shared__ float red[256];
  __shared__ float s_sc[2];
  int t = threadIdx.x;
  if (t < 16) ((int*)(ws + OFF_ZP))[t] = 0;
  float m1 = 0.f, m2 = 0.f;
  for (int i = t; i < 36864; i += 256) {
    m1 = fmaxf(m1, fabsf(w1[i]));
    m2 = fmaxf(m2, fabsf(w2[i]));
  }
  red[t] = m1; __syncthreads();
  for (int s = 128; s > 0; s >>= 1) { if (t < s) red[t] = fmaxf(red[t], red[t + s]); __syncthreads(); }
  if (t == 0) s_sc[0] = red[0] / QMAXF;
  __syncthreads();
  red[t] = m2; __syncthreads();
  for (int s = 128; s > 0; s >>= 1) { if (t < s) red[t] = fmaxf(red[t], red[t + s]); __syncthreads(); }
  if (t == 0) {
    s_sc[1] = red[0] / QMAXF;
    float* sw = (float*)(ws + OFF_SW);
    sw[0] = s_sc[0]; sw[1] = s_sc[1];
    out_tail[0] = bits[0]; out_tail[1] = f[0]; out_tail[2] = bits_out[0];
  }
  __syncthreads();
  float inv1 = 1.f / s_sc[0], inv2 = 1.f / s_sc[1];
  char* qw1 = ws + OFF_QW1;
  char* qw2 = ws + OFF_QW2;
  for (int e = t; e < 36864; e += 256) {
    int j = e & 31, lane = (e >> 5) & 31, nt = (e >> 10) & 3, tap = e >> 12;
    int oc = nt * 16 + (lane & 15);
    int ic = (j >> 3) * 16 + ((lane >= 16) ? 8 : 0) + (j & 7);
    int ky = tap / 3, kx = tap % 3;
    int widx = ((oc * 64 + ic) * 3 + ky) * 3 + kx;
    float q1 = fminf(fmaxf(rintf(w1[widx] * inv1), -QMAXF), QMAXF);
    float q2 = fminf(fmaxf(rintf(w2[widx] * inv2), -QMAXF), QMAXF);
    qw1[e] = (char)(int)q1;
    qw2[e] = (char)(int)q2;
  }
}

__global__ void k_quant1(const float* __restrict__ x, const float* alphaP,
                         char* __restrict__ xq) {
  float alpha = alphaP[0];
  float inv = QMAXF / alpha;
  long i = (long)blockIdx.x * blockDim.x + threadIdx.x;
  if (i >= NELEM) return;
  int w = (int)(i % IMG_W); long r = i / IMG_W;
  int h = (int)(r % IMG_W); r /= IMG_W;
  int c = (int)(r % 64);    int n = (int)(r / 64);
  float v = fminf(fmaxf(x[i], -alpha), alpha);
  int q = (int)rintf(v * inv);
  long pix = ((long)n * IMG_W + h) * IMG_W + w;
  xq[pix * 64 + swz(c)] = (char)q;
}

__global__ __launch_bounds__(256)
void k_conv(const char* __restrict__ xq, const char* __restrict__ qw,
            const char* __restrict__ zp,
            const float* alphaP, const float* swP, int swIdx,
            const float* preluP, int doPrelu,
            float* __restrict__ yout,
            float* __restrict__ pSum, float* __restrict__ pSq) {
  __shared__ int4 lw4[2304];
  __shared__ int4 lA4[2][TCHUNK];
  __shared__ int4 lDummy[1];
  int tid = threadIdx.x;
  int wave = tid >> 5, lane = tid & 31;
  int ocTile = wave & 3, pixPair = wave >> 2;
  int khalf = lane >> 4, mrow = lane & 15;
  int ch = ocTile * 16 + mrow;

  {
    uint32_t ldsW = (uint32_t)(uintptr_t)lw4;
#pragma unroll
    for (int k = 0; k < 9; ++k) {
      int ci = tid + k * 256;
      uint64_t g = (uint64_t)(uintptr_t)(qw + (size_t)ci * 16);
      uint32_t l = ldsW + ci * 16;
      asm volatile("global_load_async_to_lds_b128 %0, %1, off"
                   :: "v"(l), "v"(g) : "memory");
    }
  }

  uint32_t ldsDummy = (uint32_t)(uintptr_t)lDummy;

  auto stageA = [&](int seg, int buf) {
    int PB = (blockIdx.x * NSEG + seg) * SEGPIX;
    int n = (int)(PB / IMG_HW);
    int rem = (int)(PB % IMG_HW);
    int h = rem / IMG_W, w0 = rem % IMG_W;
    uint32_t ldsA = (uint32_t)(uintptr_t)(&lA4[buf][0]);
#pragma unroll
    for (int k = 0; k < 4; ++k) {
      int ci = tid + k * 256;
      bool inTile = ci < TCHUNK;
      int cc = inTile ? ci : 0;
      int e = cc >> 2, sub = cc & 3;
      int row = e / TCOLS, col = e - row * TCOLS;
      int h2 = h + row - 1, wL = w0 - 1 + col;
      bool valid = inTile && (h2 >= 0) && (h2 < IMG_W) && (wL >= 0) && (wL < IMG_W);
      const char* src = valid
          ? (xq + ((size_t)(n * IMG_W + h2) * IMG_W + wL) * 64 + sub * 16)
          : zp;
      uint32_t l = inTile ? (ldsA + ci * 16) : ldsDummy;
      asm volatile("global_load_async_to_lds_b128 %0, %1, off"
                   :: "v"(l), "v"((uint64_t)(uintptr_t)src) : "memory");
    }
  };

  stageA(0, 0);

  float mul  = (alphaP[0] / QMAXF) * swP[swIdx];
  float aPre = preluP[0];
  float s = 0.f, sq = 0.f;
  const char* lwB = (const char*)lw4;
  int colB0 = pixPair * 32 + mrow;
  int bOffBase = (ocTile * 32 + lane) * 32;

  for (int seg = 0; seg < NSEG; ++seg) {
    if (seg + 1 < NSEG) {
      stageA(seg + 1, (seg + 1) & 1);
      asm volatile("s_wait_asynccnt 0x4" ::: "memory");
    } else {
      asm volatile("s_wait_asynccnt 0x0" ::: "memory");
    }
    __syncthreads();

    const char* lAB = (const char*)(&lA4[seg & 1][0]);
    v8i acc0 = (v8i)0, acc1 = (v8i)0;
#pragma unroll
    for (int ky = 0; ky < 3; ++ky) {
#pragma unroll
      for (int kx = 0; kx < 3; ++kx) {
        int tap = ky * 3 + kx;
        union { int4 q[2]; v8i v; } bu, a0, a1;
        const int4* bp = (const int4*)(lwB + tap * 4096 + bOffBase);
        bu.q[0] = bp[0]; bu.q[1] = bp[1];
        const int4* ap0 = (const int4*)(lAB + ((ky * TCOLS + colB0 + kx) * 64 + khalf * 32));
        a0.q[0] = ap0[0]; a0.q[1] = ap0[1];
        const int4* ap1 = (const int4*)(lAB + ((ky * TCOLS + colB0 + 16 + kx) * 64 + khalf * 32));
        a1.q[0] = ap1[0]; a1.q[1] = ap1[1];
        acc0 = __builtin_amdgcn_wmma_i32_16x16x64_iu8(true, a0.v, true, bu.v, acc0, false, false);
        acc1 = __builtin_amdgcn_wmma_i32_16x16x64_iu8(true, a1.v, true, bu.v, acc1, false, false);
      }
    }

    int PB = (blockIdx.x * NSEG + seg) * SEGPIX;
    int P0 = PB + pixPair * 32;
#pragma unroll
    for (int v = 0; v < 8; ++v) {
      long p0 = (long)P0 + v + khalf * 8;
      float v0 = (float)acc0[v] * mul;
      float v1 = (float)acc1[v] * mul;
      if (doPrelu) {
        v0 = (v0 >= 0.f) ? v0 : aPre * v0;
        v1 = (v1 >= 0.f) ? v1 : aPre * v1;
      }
      yout[p0 * 64 + ch] = v0;
      yout[(p0 + 16) * 64 + ch] = v1;
      s += v0 + v1; sq += v0 * v0 + v1 * v1;
    }
    __syncthreads();
  }

  float* redS = (float*)(&lA4[0][0]);
  float* redQ = redS + 256;
  redS[tid] = s; redQ[tid] = sq;
  __syncthreads();
  if (tid < 64) {
    int oT = tid >> 4, r = tid & 15;
    int i0 = oT * 32 + r, i1 = i0 + 16, i2 = (oT + 4) * 32 + r, i3 = i2 + 16;
    pSum[(long)blockIdx.x * 64 + tid] = redS[i0] + redS[i1] + redS[i2] + redS[i3];
    pSq [(long)blockIdx.x * 64 + tid] = redQ[i0] + redQ[i1] + redQ[i2] + redQ[i3];
  }
}

__global__ void k_bn_finalize(const float* __restrict__ pSum, const float* __restrict__ pSq,
                              const float* __restrict__ gamma, const float* __restrict__ beta,
                              float* __restrict__ bnp) {
  int ch = threadIdx.x;
  if (ch >= 64) return;
  float s = 0.f, q = 0.f;
  for (int b = 0; b < NCBLK; ++b) { s += pSum[b * 64 + ch]; q += pSq[b * 64 + ch]; }
  float invN = 1.0f / (float)NPIX;
  float mean = s * invN;
  float var  = q * invN - mean * mean;
  float sc   = gamma[ch] * rsqrtf(var + BN_EPS);
  bnp[ch]      = sc;
  bnp[64 + ch] = beta[ch] - mean * sc;
}

__global__ void k_quant2(const float* __restrict__ y1, const float* __restrict__ bnp,
                         const float* alphaP, char* __restrict__ xq) {
  float alpha = alphaP[0];
  float inv = QMAXF / alpha;
  long i = (long)blockIdx.x * blockDim.x + threadIdx.x;
  if (i >= NELEM) return;
  int ch = (int)(i & 63);
  float v = y1[i] * bnp[ch] + bnp[64 + ch];
  v = fminf(fmaxf(v, -alpha), alpha);
  xq[(i >> 6) * 64 + swz(ch)] = (char)(int)rintf(v * inv);
}

__global__ void k_final(const float* __restrict__ identity, const float* __restrict__ y2,
                        const float* __restrict__ bnp, const float* shortcutP,
                        float* __restrict__ out) {
  long i = (long)blockIdx.x * blockDim.x + threadIdx.x;
  if (i >= NELEM) return;
  int w = (int)(i % IMG_W); long r = i / IMG_W;
  int h = (int)(r % IMG_W); r /= IMG_W;
  int c = (int)(r % 64);    int n = (int)(r / 64);
  long pix = ((long)n * IMG_W + h) * IMG_W + w;
  float y = y2[pix * 64 + c] * bnp[c] + bnp[64 + c];
  out[i] = identity[i] * shortcutP[0] + y;
}

extern "C" void kernel_launch(void* const* d_in, const int* in_sizes, int n_in,
                              void* d_out, int out_size, void* d_ws, size_t ws_size,
                              hipStream_t stream) {
  (void)in_sizes; (void)n_in; (void)out_size; (void)ws_size;
  const float* identity = (const float*)d_in[0];
  const float* bits     = (const float*)d_in[1];
  const float* f        = (const float*)d_in[2];
  const float* bits_out = (const float*)d_in[3];
  const float* w1       = (const float*)d_in[4];
  const float* w2       = (const float*)d_in[5];
  const float* alpha1   = (const float*)d_in[6];
  const float* alpha2   = (const float*)d_in[7];
  const float* gamma1   = (const float*)d_in[8];
  const float* beta1    = (const float*)d_in[9];
  const float* gamma2   = (const float*)d_in[10];
  const float* beta2    = (const float*)d_in[11];
  const float* prelu_a  = (const float*)d_in[12];
  const float* shortcut = (const float*)d_in[13];
  float* out = (float*)d_out;
  char*  ws  = (char*)d_ws;

  const float* swBuf = (const float*)(ws + OFF_SW);
  const char*  zp    = ws + OFF_ZP;
  float* bnp1 = (float*)(ws + OFF_BNP1);
  float* bnp2 = (float*)(ws + OFF_BNP2);
  float* pS = (float*)(ws + OFF_PSUM);
  float* pQ = (float*)(ws + OFF_PSQ);

  const int gridE = (int)((NELEM + 255) / 256);

  k_prep<<<1, 256, 0, stream>>>(w1, w2, bits, f, bits_out, ws, out + NELEM);

  k_quant1<<<gridE, 256, 0, stream>>>(identity, alpha1, ws + OFF_X1);

  k_conv<<<NCBLK, 256, 0, stream>>>(ws + OFF_X1, ws + OFF_QW1, zp,
                                    alpha1, swBuf, 0, prelu_a,  1,
                                    (float*)(ws + OFF_Y1), pS, pQ);
  k_bn_finalize<<<1, 64, 0, stream>>>(pS, pQ, gamma1, beta1, bnp1);

  k_quant2<<<gridE, 256, 0, stream>>>((const float*)(ws + OFF_Y1), bnp1, alpha2,
                                      ws + OFF_X2);

  k_conv<<<NCBLK, 256, 0, stream>>>(ws + OFF_X2, ws + OFF_QW2, zp,
                                    alpha2, swBuf, 1, prelu_a,  0,
                                    (float*)(ws + OFF_Y2), pS, pQ);
  k_bn_finalize<<<1, 64, 0, stream>>>(pS, pQ, gamma2, beta2, bnp2);

  k_final<<<gridE, 256, 0, stream>>>(identity, (const float*)(ws + OFF_Y2), bnp2,
                                     shortcut, out);
}
